// GlobalPointerHeadZeroShot_91139206021627
// MI455X (gfx1250) — matrix operand order repaired and checked by reading, not compiled, not run on hardware
//
#include <hip/hip_runtime.h>
#include <cstdint>
#include <cstddef>

#define BSZ 8
#define SEQ 2048
#define CH  768
#define ROWS (BSZ * SEQ)   // 16384

typedef __attribute__((ext_vector_type(16))) __bf16 v16bf;
typedef __attribute__((ext_vector_type(8)))  float  v8f;
typedef __attribute__((ext_vector_type(4)))  int    v4i;

union FragU { v16bf v; uint32_t u[8]; };

// ---- gfx1250 async global->LDS copy (ASYNCcnt-tracked), guarded ----
#ifdef __has_builtin
#if __has_builtin(__builtin_amdgcn_global_load_async_to_lds_b128)
#define HAVE_ASYNC_COPY 1
#endif
#if __has_builtin(__builtin_amdgcn_s_wait_asynccnt)
#define HAVE_WAIT_ASYNC 1
#endif
#endif

#if defined(HAVE_ASYNC_COPY)
#if defined(HAVE_WAIT_ASYNC)
#define WAIT_ASYNC(n) __builtin_amdgcn_s_wait_asynccnt(n)
#else
#define WAIT_ASYNC(n) asm volatile("s_wait_asynccnt %0" ::"i"(n) : "memory")
#endif
__device__ __forceinline__ void async_b128(const void* g, void* l) {
  // builtin signature (from hipcc diagnostic): (v4i AS1*, v4i AS3*, Ii, Ii)
  __builtin_amdgcn_global_load_async_to_lds_b128(
      (__attribute__((address_space(1))) v4i*)(g),
      (__attribute__((address_space(3))) v4i*)(l), 0, 0);
}
#endif

// ---- fp32 -> bf16 hi/lo split helpers (round-to-nearest-even) ----
__device__ __forceinline__ uint16_t bf_rnd(float x) {
  union { float f; uint32_t u; } t; t.f = x;
  uint32_t r = t.u + 0x7FFFu + ((t.u >> 16) & 1u);
  return (uint16_t)(r >> 16);
}
__device__ __forceinline__ float bf_up(uint16_t h) {
  union { uint32_t u; float f; } t; t.u = ((uint32_t)h) << 16;
  return t.f;
}
__device__ __forceinline__ uint32_t pack2(uint16_t a, uint16_t b) {
  return (uint32_t)a | ((uint32_t)b << 16);
}
__device__ __forceinline__ void split2(float a, float b, uint32_t& hi, uint32_t& lo) {
  uint16_t ha = bf_rnd(a), hb = bf_rnd(b);
  hi = pack2(ha, hb);
  lo = pack2(bf_rnd(a - bf_up(ha)), bf_rnd(b - bf_up(hb)));
}

// ---- WMMA fragment loaders (CDNA5 ISA §7.12.2) ----
// A 16x32 bf16: lane row = lane&15; lanes<16: V0..3=K0..7, V4..7=K16..23;
//               lanes>=16: V0..3=K8..15, V4..7=K24..31.  LDS tile: [row][16 dwords]
__device__ __forceinline__ v16bf fragA(const uint32_t* base, int lane) {
  FragU f;
  const uint32_t* p = base + (lane & 15) * 16;
  const int kb = (lane >> 4) << 2;
#pragma unroll
  for (int i = 0; i < 4; ++i) { f.u[i] = p[kb + i]; f.u[4 + i] = p[kb + 8 + i]; }
  return f.v;
}
// B 32x16 bf16: lane column = lane&15; lanes<16: V0..3=K0..7, V4..7=K16..23;
//               lanes>=16: V0..3=K8..15, V4..7=K24..31.
// LDS tile stored as [n][16 dwords].
__device__ __forceinline__ v16bf fragB(const uint32_t* base, int lane) {
  FragU f;
  const uint32_t* p = base + (lane & 15) * 16;
  const int kb = (lane >> 4) << 2;
#pragma unroll
  for (int i = 0; i < 4; ++i) { f.u[i] = p[kb + i]; f.u[4 + i] = p[kb + 8 + i]; }
  return f.v;
}

__device__ __forceinline__ v8f wmma_bf16(v16bf a, v16bf b, v8f c) {
  // (neg_a, A, neg_b, B, c_mod, C, reuse_a, reuse_b)
  return __builtin_amdgcn_wmma_f32_16x16x32_bf16(false, a, false, b, (short)0, c,
                                                 false, false);
}

// =====================================================================
// Kernel 1: x_q = F*Wq+bq, x_k = F*Wk+bk, RoPE, hi/lo bf16 split to ws.
// Grid: (CH/64, ROWS/64). 256 threads = 8 waves.
// Waves 0-3 compute a 64x64 tile of Q, waves 4-7 the same tile of K.
// =====================================================================
__global__ __launch_bounds__(256) void gp_proj_rope(
    const float* __restrict__ feat, const float* __restrict__ Wq,
    const float* __restrict__ bq,   const float* __restrict__ Wk,
    const float* __restrict__ bk,
    uint16_t* __restrict__ qhi, uint16_t* __restrict__ qlo,
    uint16_t* __restrict__ khi, uint16_t* __restrict__ klo) {
  __shared__ uint32_t sAh[64 * 16], sAl[64 * 16];
  __shared__ uint32_t sBh[2][64 * 16], sBl[2][64 * 16];

  const int tid  = threadIdx.x;
  const int lane = tid & 31, wave = tid >> 5;
  const int colBase = blockIdx.x * 64;
  const int rowBase = blockIdx.y * 64;
  const int mat = wave >> 2;  // 0 -> Q, 1 -> K
  const int wc  = wave & 3;   // 16-col strip

  v8f acc[4];
#pragma unroll
  for (int t = 0; t < 4; ++t) acc[t] = v8f{};

  for (int k0 = 0; k0 < CH; k0 += 32) {
    // prefetch next K-tile while this one is converted/consumed
    if (k0 + 32 < CH) {
      const int idx = tid * 2;
      const int pr  = idx >> 3, pkk = (idx & 7) << 2;
      __builtin_prefetch(feat + (size_t)(rowBase + pr) * CH + (k0 + 32) + pkk, 0, 3);
      const int kp = tid >> 4, nb4 = (tid & 15) << 2;
      __builtin_prefetch(Wq + (size_t)(k0 + 32 + 2 * kp) * CH + colBase + nb4, 0, 3);
      __builtin_prefetch(Wk + (size_t)(k0 + 32 + 2 * kp) * CH + colBase + nb4, 0, 3);
    }
    __syncthreads();
    // ---- stage A (feature) 64x32 fp32 -> hi/lo bf16 ----
#pragma unroll
    for (int j = 0; j < 2; ++j) {
      const int idx = tid * 2 + j;       // 512 float4 = 64x32 floats
      const int r  = idx >> 3;
      const int kk = (idx & 7) << 2;
      const float4 f = *(const float4*)(feat + (size_t)(rowBase + r) * CH + k0 + kk);
      uint32_t h0, l0, h1, l1;
      split2(f.x, f.y, h0, l0);
      split2(f.z, f.w, h1, l1);
      sAh[r * 16 + (kk >> 1)]     = h0;  sAl[r * 16 + (kk >> 1)]     = l0;
      sAh[r * 16 + (kk >> 1) + 1] = h1;  sAl[r * 16 + (kk >> 1) + 1] = l1;
    }
    // ---- stage Wq/Wk 32x64 fp32 -> hi/lo bf16, transposed to [n][k] ----
    {
      const int kp  = tid >> 4;          // k-pair 0..15
      const int nb4 = (tid & 15) << 2;   // n base 0..60
#pragma unroll
      for (int m = 0; m < 2; ++m) {
        const float* W = m ? Wk : Wq;
        const float4 r0 = *(const float4*)(W + (size_t)(k0 + 2 * kp)     * CH + colBase + nb4);
        const float4 r1 = *(const float4*)(W + (size_t)(k0 + 2 * kp + 1) * CH + colBase + nb4);
        const float* a0 = &r0.x;
        const float* a1 = &r1.x;
#pragma unroll
        for (int e = 0; e < 4; ++e) {
          uint32_t h, l;
          split2(a0[e], a1[e], h, l);
          sBh[m][(nb4 + e) * 16 + kp] = h;
          sBl[m][(nb4 + e) * 16 + kp] = l;
        }
      }
    }
    __syncthreads();

    // ---- bf16x3 emulated-fp32 WMMA ----
    const v16bf bh = fragB(&sBh[mat][wc * 16 * 16], lane);
    const v16bf bl = fragB(&sBl[mat][wc * 16 * 16], lane);
#pragma unroll
    for (int t = 0; t < 4; ++t) {
      const v16bf ah = fragA(&sAh[t * 16 * 16], lane);
      const v16bf al = fragA(&sAl[t * 16 * 16], lane);
      acc[t] = wmma_bf16(ah, bh, acc[t]);
      acc[t] = wmma_bf16(ah, bl, acc[t]);
      acc[t] = wmma_bf16(al, bh, acc[t]);
    }
  }

  // ---- epilogue: bias + RoPE + hi/lo split store ----
  const int   col   = colBase + wc * 16 + (lane & 15);
  const float bias  = (mat ? bk : bq)[col];
  const float inv   = expf((float)(col >> 1) * -0.02398526138535465f); // 1e4^(-2i/C)
  const bool  evenc = (col & 1) == 0;
  uint16_t* hdst = mat ? khi : qhi;
  uint16_t* ldst = mat ? klo : qlo;
#pragma unroll
  for (int t = 0; t < 4; ++t) {
#pragma unroll
    for (int v = 0; v < 8; ++v) {
      const int row = rowBase + t * 16 + v + ((lane >> 4) << 3);
      const int n   = row & (SEQ - 1);
      const float x  = acc[t][v] + bias;
      const float xn = __shfl_xor(x, 1, 32);      // adjacent channel partner
      float s, c;
      sincosf((float)n * inv, &s, &c);
      const float x2 = evenc ? -xn : xn;
      const float o  = x * c + x2 * s;
      const uint16_t h = bf_rnd(o);
      const uint16_t l = bf_rnd(o - bf_up(h));
      const size_t off = (size_t)row * CH + col;
      hdst[off] = h;
      ldst[off] = l;
    }
  }
}

// =====================================================================
// Kernel 2: logits[b] = Q[b] * K[b]^T (bf16x3), mask n>m, scale 1/sqrt(C).
// Grid: (SEQ/128, SEQ/128, BSZ). 256 threads = 8 waves, wave tile 64x32.
// Double-buffered LDS; async global->LDS staging when the toolchain has it.
// =====================================================================
__global__ __launch_bounds__(256) void gp_logits(
    const uint16_t* __restrict__ qhi, const uint16_t* __restrict__ qlo,
    const uint16_t* __restrict__ khi, const uint16_t* __restrict__ klo,
    float* __restrict__ out) {
  const int tid  = threadIdx.x;
  const int lane = tid & 31, wave = tid >> 5;
  const int mblk = blockIdx.x, nblk = blockIdx.y, b = blockIdx.z;
  const float scale = 0.036084391824351615f;  // 1/sqrt(768)
  float* obase = out + (size_t)b * SEQ * SEQ;

  if (nblk > mblk) {  // entire 128x128 block strictly below diagonal
    const float vmask = -1.0e10f * scale;
    const int n0 = nblk * 128, m0 = mblk * 128;
    for (int i = tid; i < 128 * 128; i += 256) {
      obase[(size_t)(n0 + (i >> 7)) * SEQ + m0 + (i & 127)] = vmask;
    }
    return;
  }

  __shared__ uint32_t sQh[2][128 * 16], sQl[2][128 * 16];
  __shared__ uint32_t sKh[2][128 * 16], sKl[2][128 * 16];

  const int rowBase = b * SEQ + nblk * 128;  // rows of Q
  const int colBase = b * SEQ + mblk * 128;  // rows of K
  const int wr = wave >> 2, wc = wave & 3;   // wave tile 64 rows x 32 cols

  v8f acc[4][2];
#pragma unroll
  for (int t = 0; t < 4; ++t)
#pragma unroll
    for (int u = 0; u < 2; ++u) acc[t][u] = v8f{};

  auto stage = [&](int buf, int k0) {
#pragma unroll
    for (int j = 0; j < 2; ++j) {
      const int idx = tid * 2 + j;       // 512 x 16B per buffer
      const int r   = idx >> 2;
      const int dw  = (idx & 3) << 2;    // dword offset within 16-dword row
      const size_t gq = (size_t)(rowBase + r) * CH + k0 + dw * 2;
      const size_t gk = (size_t)(colBase + r) * CH + k0 + dw * 2;
#if defined(HAVE_ASYNC_COPY)
      async_b128(qhi + gq, &sQh[buf][r * 16 + dw]);
      async_b128(qlo + gq, &sQl[buf][r * 16 + dw]);
      async_b128(khi + gk, &sKh[buf][r * 16 + dw]);
      async_b128(klo + gk, &sKl[buf][r * 16 + dw]);
#else
      *(uint4*)(&sQh[buf][r * 16 + dw]) = *(const uint4*)(qhi + gq);
      *(uint4*)(&sQl[buf][r * 16 + dw]) = *(const uint4*)(qlo + gq);
      *(uint4*)(&sKh[buf][r * 16 + dw]) = *(const uint4*)(khi + gk);
      *(uint4*)(&sKl[buf][r * 16 + dw]) = *(const uint4*)(klo + gk);
#endif
    }
  };

  stage(0, 0);                       // prologue fill of buffer 0
  for (int s = 0; s < CH / 32; ++s) {
    const int  cur  = s & 1;
    const bool more = (s + 1) < (CH / 32);
    if (more) stage(cur ^ 1, (s + 1) * 32);   // async fill of next buffer
#if defined(HAVE_ASYNC_COPY)
    // loads complete in order: <=8 outstanding means current tile landed
    if (more) { WAIT_ASYNC(8); } else { WAIT_ASYNC(0); }
#endif
    __syncthreads();

#pragma unroll
    for (int u = 0; u < 2; ++u) {
      const v16bf bh = fragB(&sKh[cur][(wc * 32 + u * 16) * 16], lane);
      const v16bf bl = fragB(&sKl[cur][(wc * 32 + u * 16) * 16], lane);
#pragma unroll
      for (int t = 0; t < 4; ++t) {
        const v16bf ah = fragA(&sQh[cur][(wr * 64 + t * 16) * 16], lane);
        const v16bf al = fragA(&sQl[cur][(wr * 64 + t * 16) * 16], lane);
        acc[t][u] = wmma_bf16(ah, bh, acc[t][u]);
        acc[t][u] = wmma_bf16(ah, bl, acc[t][u]);
        acc[t][u] = wmma_bf16(al, bh, acc[t][u]);
      }
    }
    __syncthreads();   // everyone done reading buf[cur] before it is refilled
  }

#pragma unroll
  for (int t = 0; t < 4; ++t)
#pragma unroll
    for (int u = 0; u < 2; ++u)
#pragma unroll
      for (int v = 0; v < 8; ++v) {
        const int n = nblk * 128 + wr * 64 + t * 16 + v + ((lane >> 4) << 3);
        const int m = mblk * 128 + wc * 32 + u * 16 + (lane & 15);
        float val = acc[t][u][v];
        if (n > m) val -= 1.0e10f;
        obase[(size_t)n * SEQ + m] = val * scale;
      }
}

extern "C" void kernel_launch(void* const* d_in, const int* in_sizes, int n_in,
                              void* d_out, int out_size, void* d_ws, size_t ws_size,
                              hipStream_t stream) {
  const float* feat = (const float*)d_in[0];
  const float* Wq   = (const float*)d_in[1];
  const float* bq   = (const float*)d_in[2];
  const float* Wk   = (const float*)d_in[3];
  const float* bk   = (const float*)d_in[4];
  float* out = (float*)d_out;

  const size_t S = (size_t)ROWS * CH;        // elements per split buffer
  uint16_t* qhi = (uint16_t*)d_ws;           // 4 buffers * 24 MB = 96 MiB total
  uint16_t* qlo = qhi + S;
  uint16_t* khi = qhi + 2 * S;
  uint16_t* klo = qhi + 3 * S;

  gp_proj_rope<<<dim3(CH / 64, ROWS / 64), 256, 0, stream>>>(
      feat, Wq, bq, Wk, bk, qhi, qlo, khi, klo);
  gp_logits<<<dim3(SEQ / 128, SEQ / 128, BSZ), 256, 0, stream>>>(
      qhi, qlo, khi, klo, out);
}
